// Model_56307021251126
// MI455X (gfx1250) — matrix operand order repaired and checked by reading, not compiled, not run on hardware
//
#include <hip/hip_runtime.h>
#include <hip/hip_bf16.h>

typedef __attribute__((ext_vector_type(16))) _Float16 v16h;
typedef __attribute__((ext_vector_type(8)))  _Float16 v8h;
typedef __attribute__((ext_vector_type(2)))  _Float16 v2h;
typedef __attribute__((ext_vector_type(8)))  float    v8f;

#define N_NODES 100000
#define N_EDGES 640000
#define D_H     128

// ---------------------------------------------------------------- utilities

__global__ __launch_bounds__(256) void zero_kernel(float* __restrict__ p, long long n) {
    long long i = (long long)blockIdx.x * 256 + threadIdx.x;
    if (i < n) p[i] = 0.0f;
}

__global__ __launch_bounds__(256) void degree_kernel(const int* __restrict__ dst,
                                                     float* __restrict__ deg, int nE) {
    int e = blockIdx.x * 256 + threadIdx.x;
    if (e < nE) atomicAdd(&deg[dst[e]], 1.0f);
}

// Pack W [K][128] (row-major, f32) -> Wt [128][K] (n-major, f16).
// One lane's WMMA-B fragment becomes 32 contiguous bytes.
__global__ __launch_bounds__(256) void pack_w_kernel(const float* __restrict__ W,
                                                     _Float16* __restrict__ Wt, int K) {
    int idx = blockIdx.x * 256 + threadIdx.x;
    if (idx >= 128 * K) return;
    int n = idx / K, k = idx - n * K;
    Wt[idx] = (_Float16)W[k * 128 + n];
}

// msum[dst[e]][c..c+3] += Hin[src[e]][c..c+3]; one thread per (edge, 4 features)
__global__ __launch_bounds__(256) void scatter_kernel(const float* __restrict__ Hin,
                                                      const int* __restrict__ src,
                                                      const int* __restrict__ dst,
                                                      float* __restrict__ msum, int nE) {
    int tid = blockIdx.x * 256 + threadIdx.x;
    int e  = tid >> 5;
    int c4 = (tid & 31) << 2;
    if (e >= nE) return;
    int s = src[e], d = dst[e];
    const float4 v = *(const float4*)(Hin + s * D_H + c4);
    float* base = msum + d * D_H + c4;
    atomicAdd(base + 0, v.x);
    atomicAdd(base + 1, v.y);
    atomicAdd(base + 2, v.z);
    atomicAdd(base + 3, v.w);
}

// A-fragment for 16x16x32 f16 WMMA: two contiguous 16B runs from an LDS row.
__device__ __forceinline__ v16h afrag(const _Float16* __restrict__ row, int kb, int hi) {
    v8h lo = *(const v8h*)(row + kb + hi * 16);
    v8h hq = *(const v8h*)(row + kb + 8 + hi * 16);
    v16h r;
    #pragma unroll
    for (int i = 0; i < 8; ++i) { r[i] = lo[i]; r[8 + i] = hq[i]; }
    return r;
}

// ------------------------------------------------- SAGE layer GEMM (WMMA)
// 32 rows/block; 8 waves, wave w owns N-tile [16w,16w+16) with 2 M-tiles.
__global__ __launch_bounds__(256) void sage_gemm_kernel(const float* __restrict__ Hin,
                                                        const float* __restrict__ Msum,
                                                        const float* __restrict__ Deg,
                                                        const _Float16* __restrict__ WsT, // [128][128] n-major f16
                                                        const _Float16* __restrict__ WnT, // [128][128] n-major f16
                                                        const float* __restrict__ Bias,
                                                        float* __restrict__ Hout) {
    __shared__ _Float16 As[32][D_H];
    __shared__ _Float16 An[32][D_H];

    const int rowBase = blockIdx.x * 32;

    // stage 32 rows of self + mean-neighbor features as f16 (packed pair stores)
    for (int i = threadIdx.x; i < 32 * 64; i += 256) {
        int r = i >> 6, c2 = (i & 63) << 1;
        int node = rowBase + r;
        float2 s = *(const float2*)(Hin + node * D_H + c2);
        float2 ms = *(const float2*)(Msum + node * D_H + c2);
        float d = Deg[node];
        d = (d < 1.0f) ? 1.0f : d;
        float inv = 1.0f / d;
        *(v2h*)&As[r][c2] = (v2h){(_Float16)s.x, (_Float16)s.y};
        *(v2h*)&An[r][c2] = (v2h){(_Float16)(ms.x * inv), (_Float16)(ms.y * inv)};
    }
    __syncthreads();

    const int lane = threadIdx.x & 31;
    const int wave = threadIdx.x >> 5;
    const int m    = lane & 15;        // A: row-in-tile / B,D: column-in-tile
    const int hi   = lane >> 4;
    const int ncol = wave * 16 + m;

    const _Float16* bsRow = WsT + ncol * D_H;   // 16 contiguous halfs per fragment
    const _Float16* bnRow = WnT + ncol * D_H;
    __builtin_prefetch(bsRow, 0, 1);            // global_prefetch_b8
    __builtin_prefetch(bnRow, 0, 1);

    v8f acc0 = {}, acc1 = {};

    #pragma unroll
    for (int kb = 0; kb < D_H; kb += 32) {      // self term
        v16h b  = *(const v16h*)(bsRow + kb + hi * 16);
        v16h a0 = afrag(&As[m][0],      kb, hi);
        v16h a1 = afrag(&As[16 + m][0], kb, hi);
        acc0 = __builtin_amdgcn_wmma_f32_16x16x32_f16(false, a0, false, b, (short)0, acc0, false, false);
        acc1 = __builtin_amdgcn_wmma_f32_16x16x32_f16(false, a1, false, b, (short)0, acc1, false, false);
    }
    #pragma unroll
    for (int kb = 0; kb < D_H; kb += 32) {      // neighbor term
        v16h b  = *(const v16h*)(bnRow + kb + hi * 16);
        v16h a0 = afrag(&An[m][0],      kb, hi);
        v16h a1 = afrag(&An[16 + m][0], kb, hi);
        acc0 = __builtin_amdgcn_wmma_f32_16x16x32_f16(false, a0, false, b, (short)0, acc0, false, false);
        acc1 = __builtin_amdgcn_wmma_f32_16x16x32_f16(false, a1, false, b, (short)0, acc1, false, false);
    }

    const float bias = Bias[ncol];
    #pragma unroll
    for (int r = 0; r < 8; ++r) {
        int row0 = rowBase + r + hi * 8;
        Hout[row0 * D_H + ncol]        = acc0[r] + bias;
        Hout[(row0 + 16) * D_H + ncol] = acc1[r] + bias;
    }
}

// ----------------------------------------------- edge-score (WMMA + reduce)
// out[e] = relu([h[s], h[d]] @ Wp1 + bp1) @ Wp2 + bp2 ; 32 edges/block
__global__ __launch_bounds__(256) void edge_score_kernel(const float* __restrict__ H,
                                                         const int* __restrict__ S,
                                                         const int* __restrict__ Dd,
                                                         const _Float16* __restrict__ Wp1T, // [128][256] n-major f16
                                                         const float* __restrict__ bp1,
                                                         const float* __restrict__ Wp2,
                                                         const float* __restrict__ bp2,
                                                         float* __restrict__ out, int nE) {
    __shared__ _Float16 Z[32][2 * D_H];
    __shared__ float score[32];

    const int eBase = blockIdx.x * 32;

    for (int i = threadIdx.x; i < 32 * 128; i += 256) {
        int r = i >> 7, c2 = (i & 127) << 1;
        int edge = eBase + r;
        const float* row = (c2 < D_H) ? (H + S[edge] * D_H) : (H + Dd[edge] * D_H - D_H);
        float2 f = *(const float2*)(row + c2);
        *(v2h*)&Z[r][c2] = (v2h){(_Float16)f.x, (_Float16)f.y};
    }
    if (threadIdx.x < 32) score[threadIdx.x] = 0.0f;
    __syncthreads();

    const int lane = threadIdx.x & 31;
    const int wave = threadIdx.x >> 5;
    const int m    = lane & 15;
    const int hi   = lane >> 4;
    const int ncol = wave * 16 + m;

    const _Float16* bRow = Wp1T + ncol * (2 * D_H);
    __builtin_prefetch(bRow, 0, 1);

    v8f acc0 = {}, acc1 = {};
    #pragma unroll
    for (int kb = 0; kb < 2 * D_H; kb += 32) {
        v16h b  = *(const v16h*)(bRow + kb + hi * 16);
        v16h a0 = afrag(&Z[m][0],      kb, hi);
        v16h a1 = afrag(&Z[16 + m][0], kb, hi);
        acc0 = __builtin_amdgcn_wmma_f32_16x16x32_f16(false, a0, false, b, (short)0, acc0, false, false);
        acc1 = __builtin_amdgcn_wmma_f32_16x16x32_f16(false, a1, false, b, (short)0, acc1, false, false);
    }

    const float b1 = bp1[ncol];
    const float w2 = Wp2[ncol];
    #pragma unroll
    for (int r = 0; r < 8; ++r) {
        float v0 = acc0[r] + b1;
        float v1 = acc1[r] + b1;
        v0 = (v0 > 0.0f) ? v0 : 0.0f;
        v1 = (v1 > 0.0f) ? v1 : 0.0f;
        atomicAdd(&score[r + hi * 8],      v0 * w2);   // ds_add_f32
        atomicAdd(&score[16 + r + hi * 8], v1 * w2);
    }
    __syncthreads();

    if (threadIdx.x < 32) {
        int edge = eBase + threadIdx.x;
        if (edge < nE) out[edge] = score[threadIdx.x] + bp2[0];
    }
}

// ---------------------------------------------------------------- launcher

extern "C" void kernel_launch(void* const* d_in, const int* in_sizes, int n_in,
                              void* d_out, int out_size, void* d_ws, size_t ws_size,
                              hipStream_t stream) {
    const float* x       = (const float*)d_in[0];
    const int*   src     = (const int*)  d_in[1];
    const int*   dst     = (const int*)  d_in[2];
    const int*   neg_src = (const int*)  d_in[3];
    const int*   neg_dst = (const int*)  d_in[4];
    const float* Ws1 = (const float*)d_in[5],  *Wn1 = (const float*)d_in[6],  *b1 = (const float*)d_in[7];
    const float* Ws2 = (const float*)d_in[8],  *Wn2 = (const float*)d_in[9],  *b2 = (const float*)d_in[10];
    const float* Ws3 = (const float*)d_in[11], *Wn3 = (const float*)d_in[12], *b3 = (const float*)d_in[13];
    const float* Wp1 = (const float*)d_in[14], *bp1 = (const float*)d_in[15];
    const float* Wp2 = (const float*)d_in[16], *bp2 = (const float*)d_in[17];

    float* ws   = (float*)d_ws;
    float* deg  = ws;                                   // N_NODES (padded to 100352)
    float* msum = ws + 100352;
    float* hA   = msum + (size_t)N_NODES * D_H;
    float* hB   = hA   + (size_t)N_NODES * D_H;
    _Float16* wt = (_Float16*)(hB + (size_t)N_NODES * D_H);  // packed f16 weights
    _Float16* WsT1 = wt;                // 128*128
    _Float16* WnT1 = wt + 16384;
    _Float16* WsT2 = wt + 32768;
    _Float16* WnT2 = wt + 49152;
    _Float16* WsT3 = wt + 65536;
    _Float16* WnT3 = wt + 81920;
    _Float16* Wp1T = wt + 98304;        // 128*256

    const long long featTotal = (long long)N_NODES * D_H;
    const int zeroFeatBlocks  = (int)((featTotal + 255) / 256);
    const int zeroDegBlocks   = (N_NODES + 255) / 256;
    const int edgeBlocks      = (N_EDGES + 255) / 256;
    const int scatBlocks      = (int)(((long long)N_EDGES * 32 + 255) / 256);  // 80000
    const int gemmBlocks      = N_NODES / 32;                                  // 3125 exact
    const int scoreBlocks     = N_EDGES / 32;                                  // 20000 exact
    const int pack128Blocks   = (128 * 128 + 255) / 256;
    const int pack256Blocks   = (128 * 256 + 255) / 256;

    // pack all weight matrices to n-major f16 (once per launch)
    pack_w_kernel<<<pack128Blocks, 256, 0, stream>>>(Ws1, WsT1, 128);
    pack_w_kernel<<<pack128Blocks, 256, 0, stream>>>(Wn1, WnT1, 128);
    pack_w_kernel<<<pack128Blocks, 256, 0, stream>>>(Ws2, WsT2, 128);
    pack_w_kernel<<<pack128Blocks, 256, 0, stream>>>(Wn2, WnT2, 128);
    pack_w_kernel<<<pack128Blocks, 256, 0, stream>>>(Ws3, WsT3, 128);
    pack_w_kernel<<<pack128Blocks, 256, 0, stream>>>(Wn3, WnT3, 128);
    pack_w_kernel<<<pack256Blocks, 256, 0, stream>>>(Wp1, Wp1T, 256);

    // degree (once)
    zero_kernel<<<zeroDegBlocks, 256, 0, stream>>>(deg, N_NODES);
    degree_kernel<<<edgeBlocks, 256, 0, stream>>>(dst, deg, N_EDGES);

    // layer 1: x -> hA
    zero_kernel<<<zeroFeatBlocks, 256, 0, stream>>>(msum, featTotal);
    scatter_kernel<<<scatBlocks, 256, 0, stream>>>(x, src, dst, msum, N_EDGES);
    sage_gemm_kernel<<<gemmBlocks, 256, 0, stream>>>(x, msum, deg, WsT1, WnT1, b1, hA);

    // layer 2: hA -> hB
    zero_kernel<<<zeroFeatBlocks, 256, 0, stream>>>(msum, featTotal);
    scatter_kernel<<<scatBlocks, 256, 0, stream>>>(hA, src, dst, msum, N_EDGES);
    sage_gemm_kernel<<<gemmBlocks, 256, 0, stream>>>(hA, msum, deg, WsT2, WnT2, b2, hB);

    // layer 3: hB -> hA
    zero_kernel<<<zeroFeatBlocks, 256, 0, stream>>>(msum, featTotal);
    scatter_kernel<<<scatBlocks, 256, 0, stream>>>(hB, src, dst, msum, N_EDGES);
    sage_gemm_kernel<<<gemmBlocks, 256, 0, stream>>>(hB, msum, deg, WsT3, WnT3, b3, hA);

    // edge scores: pos -> out[0:E], neg -> out[E:2E]
    float* out = (float*)d_out;
    edge_score_kernel<<<scoreBlocks, 256, 0, stream>>>(hA, src, dst, Wp1T, bp1, Wp2, bp2,
                                                       out, N_EDGES);
    edge_score_kernel<<<scoreBlocks, 256, 0, stream>>>(hA, neg_src, neg_dst, Wp1T, bp1, Wp2, bp2,
                                                       out + N_EDGES, N_EDGES);
}
